// HFLSTM_31293131718761
// MI455X (gfx1250) — matrix operand order repaired and checked by reading, not compiled, not run on hardware
//
#include <hip/hip_runtime.h>
#include <hip/hip_bf16.h>

// ---------------------------------------------------------------------------
// HFLSTM for MI455X (gfx1250, wave32, WMMA).
// Strategy: batch-partitioned persistent workgroups. 64 blocks x 512 threads,
// each block owns a 16-row batch tile and runs all 461 recurrent steps with
// zero inter-block sync. Per-step GEMM: gates[16,1024] = A[16,512] @ Wc^T via
// v_wmma_f32_16x16x32_bf16 (A = [relu(x@W_in^T+b_in) || h] in LDS bf16,
// Wc = [W_ih || W_hh] pre-converted to bf16, resident in L2).
// ---------------------------------------------------------------------------

typedef __attribute__((ext_vector_type(16))) __bf16        v16bf;
typedef __attribute__((ext_vector_type(8)))  float         v8f;
typedef __attribute__((ext_vector_type(4)))  unsigned int  u32x4;

#define HID    256
#define GATE4  1024          // 4*HID
#define KDIM   512           // 2*HID  (x_emb || h)
#define BT     16            // batch tile per workgroup
#define TPB    512           // 16 waves
#define NWG    64            // 1024 / BT
#define S_HIND 365
#define S_FORE 96
#define ASTR   520           // A row stride (bf16), padded for LDS banks
#define GSTR   1028          // gates row stride (f32), padded
#define WSTR   17            // W_in row stride (f32), padded

union FragBF { u32x4 u[2]; v16bf v; };

__device__ __forceinline__ float sigm_f(float x) {
    return 1.0f / (1.0f + __expf(-x));
}
__device__ __forceinline__ float tanh_f(float x) {
    x = fminf(fmaxf(x, -15.0f), 15.0f);
    float e = __expf(2.0f * x);
    return (e - 1.0f) / (e + 1.0f);
}

// One-time: Wc[j][k] = bf16( k<256 ? W_ih[j,k] : W_hh[j,k-256] ), bc = b_ih+b_hh
__global__ __launch_bounds__(256)
void hflstm_prep(const float* __restrict__ Wih, const float* __restrict__ Whh,
                 const float* __restrict__ bih, const float* __restrict__ bhh,
                 __bf16* __restrict__ Wc, float* __restrict__ bc) {
    int idx = blockIdx.x * 256 + threadIdx.x;
    if (idx < GATE4 * KDIM) {
        int j = idx >> 9, k = idx & 511;
        float v = (k < HID) ? Wih[j * HID + k] : Whh[j * HID + (k - HID)];
        Wc[idx] = (__bf16)v;
    }
    if (idx < GATE4) bc[idx] = bih[idx] + bhh[idx];
}

__global__ __launch_bounds__(TPB, 1)
void hflstm_main(const float* __restrict__ x_hind,   // [365,1024,16]
                 const float* __restrict__ xf_fore,  // [96,1024,15]
                 const float* __restrict__ W_in,     // [256,16]
                 const float* __restrict__ b_in,     // [256]
                 const __bf16* __restrict__ Wc,      // [1024,512] bf16
                 const float* __restrict__ bc,       // [1024]
                 const float* __restrict__ W_out,    // [256]
                 const float* __restrict__ b_out,    // [1]
                 float* __restrict__ out)            // [461,1024]
{
    __shared__ __align__(16) __bf16 Albf[BT][ASTR];   // A = [x_emb || h], bf16
    __shared__ float  gatesL[BT][GSTR];               // f32 gate accum / h stash
    __shared__ float  cL[BT][HID];
    __shared__ float  WinL[HID][WSTR];
    __shared__ float  binL[HID];
    __shared__ float  WoutL[HID];
    __shared__ float  bcL[GATE4];
    __shared__ float  xT[BT][16];
    __shared__ float  prevL[BT];

    const int tid  = threadIdx.x;
    const int lane = tid & 31;
    const int w    = tid >> 5;          // wave id 0..15
    const int hi   = lane >> 4;         // lane half: 0/1
    const int ln   = lane & 15;
    const int b0   = blockIdx.x * BT;
    const float bout = b_out[0];

    // ---- one-time loads / state init ----
    for (int u = tid; u < HID * 16; u += TPB) WinL[u >> 4][u & 15] = W_in[u];
    for (int u = tid; u < HID; u += TPB) { binL[u] = b_in[u]; WoutL[u] = W_out[u]; }
    for (int u = tid; u < GATE4; u += TPB) bcL[u] = bc[u];
    for (int u = tid; u < BT * HID; u += TPB) {
        int r = u >> 8, j = u & 255;
        cL[r][j] = 0.0f;
        Albf[r][HID + j] = (__bf16)0.0f;   // h = 0
    }
    __syncthreads();

    const int  ntBase = w * 4;                         // 4 N-tiles per wave
    const long wrow0  = (long)(ntBase * 16 + ln) * KDIM; // this lane's B-row base
    const int  aoff   = hi * 16;                       // A frag K sub-offset
    const int  koff   = hi * 16;                       // B frag K sub-offset

    for (int t = 0; t < S_HIND + S_FORE; ++t) {
        // ---- Phase A: stage 16x16 input tile ----
        if (t < S_HIND) {
            if (tid < BT * 16) {
                int r = tid >> 4, k = tid & 15;
                xT[r][k] = x_hind[((long)t * 1024 + b0 + r) * 16 + k];
            }
        } else {
            int tf = t - S_HIND;
            if (tid < BT * 15) {
                int r = tid / 15, k = tid % 15;
                xT[r][k] = xf_fore[((long)tf * 1024 + b0 + r) * 15 + k];
            }
            if (tid < BT) xT[tid][15] = prevL[tid];    // autoregressive feedback
        }
        __syncthreads();

        // ---- Phase B: x_emb = relu(x @ W_in^T + b_in) -> A[:,0:256] (bf16) ----
        for (int u = tid; u < BT * HID; u += TPB) {
            int r = u >> 8, j = u & 255;
            float a = binL[j];
            #pragma unroll
            for (int k = 0; k < 16; ++k) a += xT[r][k] * WinL[j][k];
            Albf[r][j] = (__bf16)fmaxf(a, 0.0f);
        }
        __syncthreads();

        // ---- Phase C: gates[16,1024] = A[16,512] x Wc^T  (WMMA bf16->f32) ----
        v8f acc0, acc1, acc2, acc3;
        {
            float q0 = bcL[(ntBase + 0) * 16 + ln];
            float q1 = bcL[(ntBase + 1) * 16 + ln];
            float q2 = bcL[(ntBase + 2) * 16 + ln];
            float q3 = bcL[(ntBase + 3) * 16 + ln];
            acc0 = (v8f){q0,q0,q0,q0,q0,q0,q0,q0};
            acc1 = (v8f){q1,q1,q1,q1,q1,q1,q1,q1};
            acc2 = (v8f){q2,q2,q2,q2,q2,q2,q2,q2};
            acc3 = (v8f){q3,q3,q3,q3,q3,q3,q3,q3};
        }
        #pragma unroll 2
        for (int k0 = 0; k0 < KDIM; k0 += 32) {
            FragBF af;   // A fragment, shared by all 4 tiles of this wave
            af.u[0] = *(const u32x4*)&Albf[ln][k0 + aoff];
            af.u[1] = *(const u32x4*)&Albf[ln][k0 + aoff + 8];
            const u32x4* p0 = (const u32x4*)(Wc + wrow0 + 0L * 16 * KDIM + k0 + koff);
            const u32x4* p1 = (const u32x4*)(Wc + wrow0 + 1L * 16 * KDIM + k0 + koff);
            const u32x4* p2 = (const u32x4*)(Wc + wrow0 + 2L * 16 * KDIM + k0 + koff);
            const u32x4* p3 = (const u32x4*)(Wc + wrow0 + 3L * 16 * KDIM + k0 + koff);
            FragBF bf0, bf1, bf2, bf3;
            bf0.u[0] = p0[0]; bf0.u[1] = p0[1];
            bf1.u[0] = p1[0]; bf1.u[1] = p1[1];
            bf2.u[0] = p2[0]; bf2.u[1] = p2[1];
            bf3.u[0] = p3[0]; bf3.u[1] = p3[1];
            acc0 = __builtin_amdgcn_wmma_f32_16x16x32_bf16(false, af.v, false, bf0.v, (short)0, acc0, false, false);
            acc1 = __builtin_amdgcn_wmma_f32_16x16x32_bf16(false, af.v, false, bf1.v, (short)0, acc1, false, false);
            acc2 = __builtin_amdgcn_wmma_f32_16x16x32_bf16(false, af.v, false, bf2.v, (short)0, acc2, false, false);
            acc3 = __builtin_amdgcn_wmma_f32_16x16x32_bf16(false, af.v, false, bf3.v, (short)0, acc3, false, false);
        }
        // C/D layout: VGPR r, lanes 0-15 -> M=r, lanes 16-31 -> M=r+8; N = lane&15
        #pragma unroll
        for (int r = 0; r < 8; ++r) {
            int row = r + 8 * hi;
            gatesL[row][(ntBase + 0) * 16 + ln] = acc0[r];
            gatesL[row][(ntBase + 1) * 16 + ln] = acc1[r];
            gatesL[row][(ntBase + 2) * 16 + ln] = acc2[r];
            gatesL[row][(ntBase + 3) * 16 + ln] = acc3[r];
        }
        __syncthreads();

        // ---- Phase D: LSTM cell elementwise (fp32) ----
        for (int u = tid; u < BT * HID; u += TPB) {
            int r = u >> 8, j = u & 255;
            float gi = gatesL[r][j];
            float gf = gatesL[r][j + 256];
            float gg = gatesL[r][j + 512];
            float go = gatesL[r][j + 768];
            float c  = sigm_f(gf) * cL[r][j] + sigm_f(gi) * tanh_f(gg);
            cL[r][j] = c;
            float h  = sigm_f(go) * tanh_f(c);
            gatesL[r][j]     = h;            // stash f32 h (only this thread reads slot)
            Albf[r][HID + j] = (__bf16)h;    // h for next step's GEMM
        }
        __syncthreads();

        // ---- Phase E: y = h @ W_out^T + b_out (one wave per batch row) ----
        {
            float p = 0.0f;
            #pragma unroll
            for (int j = lane; j < HID; j += 32) p += gatesL[w][j] * WoutL[j];
            #pragma unroll
            for (int off = 16; off; off >>= 1) p += __shfl_xor(p, off, 32);
            if (lane == 0) {
                float y = p + bout;
                out[(long)t * 1024 + b0 + w] = y;
                prevL[w] = y;
            }
        }
        __syncthreads();
    }
}

extern "C" void kernel_launch(void* const* d_in, const int* in_sizes, int n_in,
                              void* d_out, int out_size, void* d_ws, size_t ws_size,
                              hipStream_t stream) {
    const float* x_hind = (const float*)d_in[0];
    const float* xf     = (const float*)d_in[1];
    const float* W_in   = (const float*)d_in[2];
    const float* b_in   = (const float*)d_in[3];
    const float* W_ih   = (const float*)d_in[4];
    const float* W_hh   = (const float*)d_in[5];
    const float* b_ih   = (const float*)d_in[6];
    const float* b_hh   = (const float*)d_in[7];
    const float* W_out  = (const float*)d_in[8];
    const float* b_out  = (const float*)d_in[9];
    float* out = (float*)d_out;

    // Workspace: Wc (1 MB bf16) then bc (4 KB f32)
    __bf16* Wc = (__bf16*)d_ws;
    float*  bc = (float*)((char*)d_ws + (size_t)GATE4 * KDIM * sizeof(__bf16));

    hflstm_prep<<<(GATE4 * KDIM + 255) / 256, 256, 0, stream>>>(W_ih, W_hh, b_ih, b_hh, Wc, bc);
    hflstm_main<<<NWG, TPB, 0, stream>>>(x_hind, xf, W_in, b_in, Wc, bc, W_out, b_out, out);
}
